// gwnet_24756191494285
// MI455X (gfx1250) — matrix operand order repaired and checked by reading, not compiled, not run on hardware
//
#include <hip/hip_runtime.h>

// ---------------------------------------------------------------------------
// Graph WaveNet inference for MI455X (gfx1250, wave32, WMMA).
// One generic bf16 WMMA GEMM (f32 accumulate). All operands arranged so every
// fragment load is an unconditional 16-byte global_load_b128:
//   * A: row-major, K multiple of 32, rows 16B-aligned (two uint4 per step)
//   * B: row-major (N x K), contiguous K (two uint4 per step)
// GCN's node contraction uses zero-padded K=224 transposed operands that are
// produced by the previous stage (transposed-store epilogue / k_gated).
// ---------------------------------------------------------------------------

typedef __attribute__((ext_vector_type(16))) __bf16 v16bf;
typedef __attribute__((ext_vector_type(8)))  float  v8f;

struct Q2 { uint4 a, b; };   // 32 bytes == one 16-element bf16 fragment

#define GF_RELU 1
#define GF_ACC  2

// A row address: abase(m) = (m/rowChunkA)*chunkStrideA + (m%rowChunkA)*rowStrideA
// k-step remap (dilated conv taps): ka = kt < kSplit ? kt : kOff2 + (kt-kSplit)
//   (kSplit is 32-aligned so a whole 32-wide k-step is on one side)
__global__ __launch_bounds__(128) void wmma_gemm(
    const __bf16* __restrict__ A, long aBatchStride,
    int rowChunkA, long chunkStrideA, long rowStrideA,
    int kSplit, int kOff2,
    const __bf16* __restrict__ Bm, long bBatchStride, long sBn,
    const float* __restrict__ bias,
    float* __restrict__ Cf, __bf16* __restrict__ Cb, long cBatchStride, long ldc,
    __bf16* __restrict__ CbT, long ldcT, long cTBatchStride,
    int M, int N, int K, int flags)
{
  const int lane = threadIdx.x;       // 0..31
  const int half = lane >> 4;         // 0/1
  const int l16  = lane & 15;
  const int mTile = blockIdx.y * blockDim.y + threadIdx.y;
  const int m0 = mTile * 16;
  if (m0 >= M) return;                // uniform per wave
  const int n0 = blockIdx.x * 16;
  const int z  = blockIdx.z;

  const __bf16* Ab = A  + (long)z * aBatchStride;
  const __bf16* Bb = Bm + (long)z * bBatchStride;

  // per-lane A row (clamped; stores are masked, extra row reads are benign)
  const int  arow = m0 + l16;
  const int  arc  = (arow < M) ? arow : (M - 1);
  const long abase = (long)(arc / rowChunkA) * chunkStrideA
                   + (long)(arc % rowChunkA) * rowStrideA;
  const __bf16* ArowP = Ab + abase;

  // per-lane B column (clamped)
  const int  ncol = n0 + l16;
  const bool nval = ncol < N;
  const int  ncc  = nval ? ncol : (N - 1);
  const __bf16* BrowP = Bb + (long)ncc * sBn;

  const int aOff = half << 3;   // 8*half
  const int bOff = half << 3;   // 8*half

  v8f acc = {};
  for (int kt = 0; kt < K; kt += 32) {
    long ka = (kt < kSplit) ? (long)kt : ((long)kOff2 + (kt - kSplit));
    const __bf16* Ap = ArowP + ka;
    Q2 qa, qb;
    qa.a = *(const uint4*)(Ap + aOff);          // k = ka + 8*half + 0..7
    qa.b = *(const uint4*)(Ap + 16 + aOff);     // k = ka + 16 + 8*half + 0..7
    const __bf16* Bp = BrowP + kt + bOff;
    qb.a = *(const uint4*)(Bp);                 // k = kt + 8*half + 0..7
    qb.b = *(const uint4*)(Bp + 16);            // k = kt + 16 + 8*half + 0..7
    v16bf av = __builtin_bit_cast(v16bf, qa);
    v16bf bv = __builtin_bit_cast(v16bf, qb);
    acc = __builtin_amdgcn_wmma_f32_16x16x32_bf16(
        false, av, false, bv, (short)0, acc, false, false);
  }

  const long cb  = (long)z * cBatchStride;
  const long ctb = (long)z * cTBatchStride;
#pragma unroll
  for (int r = 0; r < 8; ++r) {
    int mrow = m0 + r + (half << 3);   // D layout: VGPR r -> M = r + 8*half
    if (mrow < M && nval) {
      float v = acc[r];
      if (bias) v += bias[ncol];
      if (flags & GF_RELU) v = fmaxf(v, 0.0f);
      if (flags & GF_ACC)      Cf[cb + (long)mrow * ldc + ncol] += v;
      else if (Cf)             Cf[cb + (long)mrow * ldc + ncol]  = v;
      if (Cb)                  Cb[cb + (long)mrow * ldc + ncol]  = (__bf16)v;
      if (CbT)                 CbT[ctb + (long)ncol * ldcT + mrow] = (__bf16)v;
    }
  }
}

// ----------------------------- pointwise kernels ---------------------------

// adjacency: adpT[w*224+v] = softmax_w(relu(E1[v,:] @ E2[:,w])), cols 207.. stay 0
__global__ void k_adp(const float* __restrict__ E1, const float* __restrict__ E2,
                      __bf16* __restrict__ adpT)
{
  const int Nn = 207;
  __shared__ float sc[207];
  __shared__ float red[256];
  int v = blockIdx.x, tid = threadIdx.x;
  for (int w = tid; w < Nn; w += 256) {
    float s = 0.f;
    for (int k = 0; k < 10; ++k) s += E1[v * 10 + k] * E2[k * Nn + w];
    sc[w] = fmaxf(s, 0.0f);
  }
  __syncthreads();
  float mx = -1e30f;
  for (int w = tid; w < Nn; w += 256) mx = fmaxf(mx, sc[w]);
  red[tid] = mx; __syncthreads();
  for (int s = 128; s > 0; s >>= 1) { if (tid < s) red[tid] = fmaxf(red[tid], red[tid + s]); __syncthreads(); }
  mx = red[0]; __syncthreads();
  float sum = 0.f;
  for (int w = tid; w < Nn; w += 256) { float e = __expf(sc[w] - mx); sc[w] = e; sum += e; }
  red[tid] = sum; __syncthreads();
  for (int s = 128; s > 0; s >>= 1) { if (tid < s) red[tid] += red[tid + s]; __syncthreads(); }
  float inv = 1.0f / red[0];
  for (int w = tid; w < Nn; w += 256) adpT[(long)w * 224 + v] = (__bf16)(sc[w] * inv);
}

// start conv: x (B,2,N,T) f32 -> h0 (B,N,T,32) bf16
__global__ void k_start(const float* __restrict__ x, const float* __restrict__ w,
                        const float* __restrict__ b, __bf16* __restrict__ h0, long total)
{
  long i = (long)blockIdx.x * blockDim.x + threadIdx.x;
  if (i >= total) return;
  int  c = (int)(i & 31);
  long m = i >> 5;
  int  t = (int)(m % 168);
  long bn = m / 168;
  int  n = (int)(bn % 207), bi = (int)(bn / 207);
  float x0 = x[((long)(bi * 2 + 0) * 207 + n) * 168 + t];
  float x1 = x[((long)(bi * 2 + 1) * 207 + n) * 168 + t];
  h0[i] = (__bf16)(w[c * 2 + 0] * x0 + w[c * 2 + 1] * x1 + b[c]);
}

// pack (L,DC,RC,2) -> (L,DC,64) bf16: [tap0 ch0..31 | tap1 ch0..31]
__global__ void k_pack2(const float* __restrict__ w, __bf16* __restrict__ out, int total)
{
  int i = blockIdx.x * blockDim.x + threadIdx.x;
  if (i >= total) return;
  int j  = i & 1;
  int ii = (i >> 1) & 31;
  int lo = i >> 6;                       // l*32 + o
  out[(long)lo * 64 + j * 32 + ii] = (__bf16)w[i];
}

__global__ void k_cast(const float* __restrict__ in, __bf16* __restrict__ out, long n, int relu)
{
  long i = (long)blockIdx.x * blockDim.x + threadIdx.x;
  if (i >= n) return;
  float v = in[i];
  if (relu) v = fmaxf(v, 0.0f);
  out[i] = (__bf16)v;
}

// gated = tanh(f)*sigmoid(g): writes (B,N,To,32) and transposed (B, To*32, 224)
__global__ void k_gated(const float* __restrict__ f, const float* __restrict__ g,
                        __bf16* __restrict__ out, __bf16* __restrict__ outT,
                        int Tout, long tBatchStride, long n)
{
  long i = (long)blockIdx.x * blockDim.x + threadIdx.x;
  if (i >= n) return;
  float t = tanhf(f[i]);
  float s = 1.0f / (1.0f + __expf(-g[i]));
  __bf16 v = (__bf16)(t * s);
  out[i] = v;
  int  c = (int)(i & 31);
  long m = i >> 5;
  int  tt = (int)(m % Tout);
  long bn = m / Tout;
  int  nd = (int)(bn % 207);
  long bi = bn / 207;
  outT[bi * tBatchStride + (long)(tt * 32 + c) * 224 + nd] = v;
}

// h_next = (gcn_out + residual[last Tout]) * gamma/sqrt(1+eps) + beta -> bf16
__global__ void k_bnres(const float* __restrict__ gcn, const __bf16* __restrict__ hres,
                        const float* __restrict__ gamma, const float* __restrict__ beta,
                        __bf16* __restrict__ hout, int Tout, int Tin, long total)
{
  long i = (long)blockIdx.x * blockDim.x + threadIdx.x;
  if (i >= total) return;
  int  c = (int)(i & 31);
  long m = i >> 5;
  int  t = (int)(m % Tout);
  long bn = m / Tout;
  float r = (float)hres[(bn * Tin + t + (Tin - Tout)) * 32 + c];
  float v = gcn[i] + r;
  float scale = gamma[c] * rsqrtf(1.0f + 1e-5f);
  hout[i] = (__bf16)(v * scale + beta[c]);
}

// final linear over time: e2 (B,N,159,12) -> out (B,12,N)
__global__ void k_final(const float* __restrict__ e2, const float* __restrict__ lw,
                        const float* __restrict__ lb, float* __restrict__ out)
{
  int i = blockIdx.x * blockDim.x + threadIdx.x;
  if (i >= 16 * 12 * 207) return;
  int n = i % 207;
  int o = (i / 207) % 12;
  int b = i / (207 * 12);
  const float* p = e2 + ((long)(b * 207 + n) * 159) * 12 + o;
  float s = 0.f;
  for (int t = 0; t < 159; ++t) s += p[(long)t * 12] * lw[t];
  out[i] = s + lb[0];
}

// ---------------------------------------------------------------------------

static inline void launch_gemm(hipStream_t st,
    const __bf16* A, long aBatch, int rowChunk, long chunkStride, long rowStride,
    int kSplit, int kOff2,
    const __bf16* Bm, long bBatch, long sBn,
    const float* bias, float* Cf, __bf16* Cb, long cBatch, long ldc,
    __bf16* CbT, long ldcT, long cTBatch,
    int M, int Nn, int K, int flags, int batches)
{
  dim3 blk(32, 4, 1);
  int mTiles = (M + 15) / 16;
  dim3 grd((Nn + 15) / 16, (mTiles + 3) / 4, batches);
  wmma_gemm<<<grd, blk, 0, st>>>(A, aBatch, rowChunk, chunkStride, rowStride,
                                 kSplit, kOff2, Bm, bBatch, sBn, bias,
                                 Cf, Cb, cBatch, ldc, CbT, ldcT, cTBatch,
                                 M, Nn, K, flags);
}

extern "C" void kernel_launch(void* const* d_in, const int* in_sizes, int n_in,
                              void* d_out, int out_size, void* d_ws, size_t ws_size,
                              hipStream_t stream)
{
  (void)in_sizes; (void)n_in; (void)out_size; (void)ws_size;
  const float* x        = (const float*)d_in[0];
  const float* start_w  = (const float*)d_in[1];
  const float* start_b  = (const float*)d_in[2];
  const float* nv1      = (const float*)d_in[3];
  const float* nv2      = (const float*)d_in[4];
  const float* filter_w = (const float*)d_in[5];
  const float* filter_b = (const float*)d_in[6];
  const float* gate_w   = (const float*)d_in[7];
  const float* gate_b   = (const float*)d_in[8];
  const float* skip_w   = (const float*)d_in[9];
  const float* skip_b   = (const float*)d_in[10];
  const float* gcn_w    = (const float*)d_in[11];
  const float* gcn_b    = (const float*)d_in[12];
  const float* bn_gamma = (const float*)d_in[13];
  const float* bn_beta  = (const float*)d_in[14];
  const float* end1_w   = (const float*)d_in[15];
  const float* end1_b   = (const float*)d_in[16];
  const float* end2_w   = (const float*)d_in[17];
  const float* end2_b   = (const float*)d_in[18];
  const float* lin_w    = (const float*)d_in[19];
  const float* lin_b    = (const float*)d_in[20];
  float* outp           = (float*)d_out;

  const int Bv = 16, Nn = 207, Tt = 168, TOUT = 159, KP = 224;
  const int dil[6] = {1, 2, 1, 2, 1, 2};
  const int BIG = 1 << 30;
  const long TSTRIDE = (long)Tt * 32 * KP;   // fixed per-batch stride of *T buffers

  // workspace carve-up (all offsets 256B aligned)
  char* W = (char*)d_ws;
  size_t off = 0;
  auto alloc = [&](size_t bytes) -> size_t {
    size_t r = off; off = (off + bytes + 255) & ~(size_t)255; return r;
  };
  __bf16* adpT   = (__bf16*)(W + alloc((size_t)207 * KP * 2));
  __bf16* wf     = (__bf16*)(W + alloc((size_t)6 * 32 * 64 * 2));
  __bf16* wg     = (__bf16*)(W + alloc((size_t)6 * 32 * 64 * 2));
  __bf16* wskip  = (__bf16*)(W + alloc((size_t)6 * 256 * 32 * 2));
  __bf16* wgcn   = (__bf16*)(W + alloc((size_t)6 * 32 * 128 * 2));
  __bf16* we1    = (__bf16*)(W + alloc((size_t)512 * 256 * 2));
  __bf16* we2    = (__bf16*)(W + alloc((size_t)12 * 512 * 2));
  const size_t actMax = (size_t)Bv * Nn * Tt * 32;   // elements
  __bf16* hA     = (__bf16*)(W + alloc(actMax * 2));
  __bf16* hB     = (__bf16*)(W + alloc(actMax * 2));
  __bf16* gated  = (__bf16*)(W + alloc(actMax * 2));
  __bf16* x1b    = (__bf16*)(W + alloc(actMax * 2));
  __bf16* x2b    = (__bf16*)(W + alloc(actMax * 2));
  __bf16* x3b    = (__bf16*)(W + alloc(actMax * 2));
  const size_t tElems = (size_t)Bv * TSTRIDE;        // transposed (K=224-padded)
  __bf16* gatedT = (__bf16*)(W + alloc(tElems * 2));
  __bf16* x1T    = (__bf16*)(W + alloc(tElems * 2));
  __bf16* x2T    = (__bf16*)(W + alloc(tElems * 2));
  float*  S1     = (float*)(W + alloc(actMax * 4));
  float*  S2     = (float*)(W + alloc(actMax * 4));
  const size_t skipElems = (size_t)Bv * Nn * TOUT * 256;
  float*  skipAcc  = (float*)(W + alloc(skipElems * 4));   // reused as e1 bf16 (x512)
  __bf16* skipRelu = (__bf16*)(W + alloc(skipElems * 2));
  float*  e2buf    = (float*)(W + alloc((size_t)Bv * Nn * TOUT * 12 * 4));
  __bf16* e1buf    = (__bf16*)skipAcc;                     // 512 bf16 == 256 f32 bytes

  // ---- prep ----
  hipMemsetAsync(adpT,   0, (size_t)207 * KP * 2, stream);  // zero K-pad cols
  hipMemsetAsync(gatedT, 0, tElems * 2, stream);            // zero K-pad cols
  hipMemsetAsync(x1T,    0, tElems * 2, stream);
  hipMemsetAsync(x2T,    0, tElems * 2, stream);
  hipMemsetAsync(skipAcc, 0, skipElems * 4, stream);
  k_adp<<<207, 256, 0, stream>>>(nv1, nv2, adpT);
  {
    long tot = (long)actMax;
    k_start<<<(unsigned)((tot + 255) / 256), 256, 0, stream>>>(x, start_w, start_b, hA, tot);
  }
  k_pack2<<<(12288 + 255) / 256, 256, 0, stream>>>(filter_w, wf, 12288);
  k_pack2<<<(12288 + 255) / 256, 256, 0, stream>>>(gate_w,   wg, 12288);
  k_cast<<<(49152  + 255) / 256, 256, 0, stream>>>(skip_w, wskip, 49152, 0);
  k_cast<<<(24576  + 255) / 256, 256, 0, stream>>>(gcn_w,  wgcn,  24576, 0);
  k_cast<<<(131072 + 255) / 256, 256, 0, stream>>>(end1_w, we1, 131072, 0);
  k_cast<<<(6144   + 255) / 256, 256, 0, stream>>>(end2_w, we2, 6144, 0);

  // ---- residual blocks ----
  int Tc = Tt;
  __bf16* hcur = hA;
  __bf16* hnxt = hB;
  for (int i = 0; i < 6; ++i) {
    const int d = dil[i];
    const int To = Tc - d;
    const int Mc = Bv * Nn * To;          // positions this layer
    const long nAct = (long)Mc * 32;
    const long gb = (long)Nn * To * 32;   // per-batch stride, normal layout

    // filter / gate dilated convs: K=64 via k-step remap (tap1 at +d*32)
    launch_gemm(stream, hcur, 0, To, (long)Tc * 32, 32, 32, d * 32,
                wf + (size_t)i * 2048, 0, 64, filter_b + i * 32,
                S1, nullptr, 0, 32, nullptr, 0, 0, Mc, 32, 64, 0, 1);
    launch_gemm(stream, hcur, 0, To, (long)Tc * 32, 32, 32, d * 32,
                wg + (size_t)i * 2048, 0, 64, gate_b + i * 32,
                S2, nullptr, 0, 32, nullptr, 0, 0, Mc, 32, 64, 0, 1);
    k_gated<<<(unsigned)((nAct + 255) / 256), 256, 0, stream>>>(
        S1, S2, gated, gatedT, To, TSTRIDE, nAct);

    // skip conv over last 159 timesteps, accumulated into skipAcc (f32 +=)
    launch_gemm(stream, gated + (size_t)(To - TOUT) * 32, 0, TOUT, (long)To * 32, 32,
                BIG, 0, wskip + (size_t)i * 256 * 32, 0, 32, skip_b + i * 256,
                skipAcc, nullptr, 0, 256, nullptr, 0, 0,
                Bv * Nn * TOUT, 256, 32, GF_ACC, 1);

    // GCN diffusion x{k+1}[b] = adpT (207x224) @ x{k}T (224 x To*32), batched.
    // Each stage stores both the normal (w,t,c) layout and the 224-padded
    // transposed layout consumed as B by the next stage.
    launch_gemm(stream, adpT, 0, 207, 0, KP, BIG, 0,
                gatedT, TSTRIDE, KP, nullptr,
                nullptr, x1b, gb, (long)To * 32, x1T, KP, TSTRIDE,
                207, To * 32, KP, 0, Bv);
    launch_gemm(stream, adpT, 0, 207, 0, KP, BIG, 0,
                x1T, TSTRIDE, KP, nullptr,
                nullptr, x2b, gb, (long)To * 32, x2T, KP, TSTRIDE,
                207, To * 32, KP, 0, Bv);
    launch_gemm(stream, adpT, 0, 207, 0, KP, BIG, 0,
                x2T, TSTRIDE, KP, nullptr,
                nullptr, x3b, gb, (long)To * 32, nullptr, 0, 0,
                207, To * 32, KP, 0, Bv);

    // GCN 1x1 conv (K=128) as 4 accumulating K=32 chunks: [gated|x1|x2|x3]
    const __bf16* chunks[4] = {gated, x1b, x2b, x3b};
    for (int c = 0; c < 4; ++c) {
      launch_gemm(stream, chunks[c], 0, Mc, 0, 32, BIG, 0,
                  wgcn + (size_t)i * 4096 + c * 32, 0, 128,
                  (c == 0) ? (gcn_b + i * 32) : nullptr,
                  S1, nullptr, 0, 32, nullptr, 0, 0,
                  Mc, 32, 32, (c == 0) ? 0 : GF_ACC, 1);
    }

    // residual add + BatchNorm (eval) -> next h (bf16)
    k_bnres<<<(unsigned)((nAct + 255) / 256), 256, 0, stream>>>(
        S1, hcur, bn_gamma + i * 32, bn_beta + i * 32, hnxt, To, Tc, nAct);

    __bf16* tmp = hcur; hcur = hnxt; hnxt = tmp;
    Tc = To;
  }

  // ---- head ----
  const int Mend = Bv * Nn * TOUT;        // 526608 positions
  {
    long n = (long)Mend * 256;
    k_cast<<<(unsigned)((n + 255) / 256), 256, 0, stream>>>(skipAcc, skipRelu, n, 1);
  }
  // end1: 256 -> 512, relu, bf16 out (reuses skipAcc bytes)
  launch_gemm(stream, skipRelu, 0, Mend, 0, 256, BIG, 0,
              we1, 0, 256, end1_b,
              nullptr, e1buf, 0, 512, nullptr, 0, 0, Mend, 512, 256, GF_RELU, 1);
  // end2: 512 -> 12, f32 out
  launch_gemm(stream, e1buf, 0, Mend, 0, 512, BIG, 0,
              we2, 0, 512, end2_b,
              e2buf, nullptr, 0, 12, nullptr, 0, 0, Mend, 12, 512, 0, 1);
  // linear over time + bias -> (B,12,N,1)
  k_final<<<(16 * 12 * 207 + 255) / 256, 256, 0, stream>>>(e2buf, lin_w, lin_b, outp);
}
